// Breaker_25056839204886
// MI455X (gfx1250) — compile-verified, matrix operand order checked by reading, not run on hardware
//
#include <hip/hip_runtime.h>

typedef __attribute__((ext_vector_type(16))) __bf16       v16bf;
typedef __attribute__((ext_vector_type(8)))  float        v8f;
typedef __attribute__((ext_vector_type(8)))  unsigned int v8u;

#define WMMA_BF16(a, b, c) \
  __builtin_amdgcn_wmma_f32_16x16x32_bf16(false, (a), false, (b), (short)0, (c), false, false)

// ---------------------------------------------------------------------------
// fp32 -> bf16 round-to-nearest-even, returned as raw u16
__device__ __forceinline__ unsigned short f2bf(float f) {
  unsigned int u = __builtin_bit_cast(unsigned int, f);
  u += 0x7fffu + ((u >> 16) & 1u);
  return (unsigned short)(u >> 16);
}

// Load one 16x32 bf16 WMMA fragment (A or B role) from a K-contiguous matrix.
// Storage: row-major [row][rowElems] of raw-bf16 u16, rowElems even, kc even.
// Per ISA 7.12.2 (16-bit A 16x32): lane<16 -> K offsets {2v + 8*[v>=4]},
// lane>=16 -> same +8; each VGPR = packed pair (k, k+1) = one b32 load.
__device__ __forceinline__ v16bf frag_ld(const unsigned short* p0, int rowElems,
                                         int row, int kc, int lane) {
  const int hi = lane >> 4;
  const unsigned short* p = p0 + (size_t)row * rowElems + kc + (hi ? 8 : 0);
  v8u u;
#pragma unroll
  for (int v = 0; v < 8; ++v) {
    const int kb = 2 * v + (v >= 4 ? 8 : 0);
    u[v] = *(const unsigned int*)(p + kb);
  }
  return __builtin_bit_cast(v16bf, u);
}

// ---------------------------------------------------------------------------
// Constants
#define NB      65536
#define NFIELD  40
#define EMBED   5
#define K1      200   // F*E
#define K1P     224   // padded to 7 chunks of 32
#define H1      256
#define DREP    64
#define XK      69    // DREP + EMBED
#define XKP     96    // padded to 3 chunks of 32
#define NC      16
#define T1N     32
#define T2N     10
#define T2NP    16

// ---------------------------------------------------------------------------
// Kernel 0: weight conversion/padding into WMMA-friendly bf16 [N][Kpad] layout
__global__ __launch_bounds__(256) void bk_prep(
    const float* __restrict__ w_rep1, const float* __restrict__ w_rep2,
    const float* __restrict__ tw1,    const float* __restrict__ tw2,
    unsigned short* __restrict__ W1p, unsigned short* __restrict__ W2p,
    unsigned short* __restrict__ T1p, unsigned short* __restrict__ T2p) {
  int idx = blockIdx.x * 256 + threadIdx.x;
  if (idx < H1 * K1P) {                       // W1p[n][k] = w_rep1[k][n]
    int n = idx / K1P, k = idx - n * K1P;
    W1p[idx] = f2bf(k < K1 ? w_rep1[k * H1 + n] : 0.f);
    return;
  }
  idx -= H1 * K1P;
  if (idx < DREP * H1) {                      // W2p[n][k] = w_rep2[k][n]
    int n = idx / H1, k = idx - n * H1;
    W2p[idx] = f2bf(w_rep2[k * DREP + n]);
    return;
  }
  idx -= DREP * H1;
  if (idx < NC * T1N * XKP) {                 // T1p[c][n][k] = tw1[c][k][n]
    int c = idx / (T1N * XKP), r = idx - c * T1N * XKP;
    int n = r / XKP, k = r - n * XKP;
    T1p[idx] = f2bf(k < XK ? tw1[(c * XK + k) * T1N + n] : 0.f);
    return;
  }
  idx -= NC * T1N * XKP;
  if (idx < NC * T2NP * T1N) {                // T2p[c][n][k] = tw2[c][k][n]
    int c = idx / (T2NP * T1N), r = idx - c * T2NP * T1N;
    int n = r / T1N, k = r - n * T1N;
    T2p[idx] = f2bf(n < T2N ? tw2[(c * T1N + k) * T2N + n] : 0.f);
  }
}

// ---------------------------------------------------------------------------
// Kernel 1: embedding gather + GEMM1(200->256,ReLU) + GEMM2(256->64)
// Block = 256 threads (8 waves), 32 batch rows per block.
__global__ __launch_bounds__(256) void bk_user(
    const int*   __restrict__ u_feat,     const int*   __restrict__ i_feat,
    const float* __restrict__ user_table, const float* __restrict__ b_rep1,
    const float* __restrict__ b_rep2,     const float* __restrict__ item_table,
    const unsigned short* __restrict__ W1p, const unsigned short* __restrict__ W2p,
    unsigned short* __restrict__ xbuf, float* __restrict__ urep) {
  __shared__ unsigned short A[32 * K1P];   // gathered embeddings, bf16
  __shared__ unsigned short H[32 * H1];    // hidden activations, bf16
  const int b0  = blockIdx.x * 32;
  const int tid = threadIdx.x;

  // cooperative gather: 32 rows x 224 K (zeros beyond 200)
  for (int idx = tid; idx < 32 * K1P; idx += 256) {
    int row = idx / K1P, k = idx - row * K1P;
    float val = 0.f;
    if (k < K1) {
      int f = k / EMBED, e = k - f * EMBED;
      int feat = u_feat[(size_t)(b0 + row) * NFIELD + f];
      val = user_table[(size_t)feat * EMBED + e];
    }
    A[idx] = f2bf(val);
  }
  __syncthreads();

  const int wave = tid >> 5, lane = tid & 31;
  const int hi = lane >> 4, ln = lane & 15;
  const int m0 = (wave >> 2) * 16;        // row slab: 2 slabs of 16
  const int nb = (wave & 3) * 64;         // GEMM1 col span: 4 tiles of 16

  // ---- GEMM1: [32 x 224] x [224 x 256], bias-initialized accumulators
  v8f acc[4];
#pragma unroll
  for (int t = 0; t < 4; ++t) {
    float bias = b_rep1[nb + t * 16 + ln];
#pragma unroll
    for (int r = 0; r < 8; ++r) acc[t][r] = bias;
  }
  for (int kc = 0; kc < K1P; kc += 32) {
    v16bf af = frag_ld(A, K1P, m0 + ln, kc, lane);
#pragma unroll
    for (int t = 0; t < 4; ++t) {
      v16bf bf = frag_ld(W1p, K1P, nb + t * 16 + ln, kc, lane);
      acc[t] = WMMA_BF16(af, bf, acc[t]);
    }
  }
  // ReLU -> bf16 -> LDS (D-layout: m = r + 8*hi, n = ln)
#pragma unroll
  for (int t = 0; t < 4; ++t)
#pragma unroll
    for (int r = 0; r < 8; ++r) {
      float v = acc[t][r];
      v = v > 0.f ? v : 0.f;
      H[(m0 + r + 8 * hi) * H1 + nb + t * 16 + ln] = f2bf(v);
    }
  __syncthreads();

  // ---- GEMM2: [32 x 256] x [256 x 64]
  const int nb2 = (wave & 3) * 16;
  v8f acc2;
  {
    float bias = b_rep2[nb2 + ln];
#pragma unroll
    for (int r = 0; r < 8; ++r) acc2[r] = bias;
  }
  for (int kc = 0; kc < H1; kc += 32) {
    v16bf af = frag_ld(H,   H1, m0 + ln,  kc, lane);
    v16bf bf = frag_ld(W2p, H1, nb2 + ln, kc, lane);
    acc2 = WMMA_BF16(af, bf, acc2);
  }
#pragma unroll
  for (int r = 0; r < 8; ++r) {
    int m = m0 + r + 8 * hi, n = nb2 + ln;
    float v = acc2[r];
    urep[(size_t)(b0 + m) * DREP + n] = v;           // fp32 for q-assignment
    xbuf[(size_t)(b0 + m) * XKP  + n] = f2bf(v);     // bf16 for towers
  }
  // item embedding + zero-pad into x cols [64,96)
  for (int idx = tid; idx < 32 * 32; idx += 256) {
    int row = idx >> 5, e = idx & 31;
    float v = (e < EMBED) ? item_table[i_feat[b0 + row] * EMBED + e] : 0.f;
    xbuf[(size_t)(b0 + row) * XKP + DREP + e] = f2bf(v);
  }
}

// ---------------------------------------------------------------------------
// Kernel 2: 16 cluster towers. Block = 256 threads, 16 rows/block,
// wave w handles clusters w and w+8.
__global__ __launch_bounds__(256) void bk_tower(
    const unsigned short* __restrict__ xbuf,
    const unsigned short* __restrict__ T1p, const unsigned short* __restrict__ T2p,
    const float* __restrict__ tb1, const float* __restrict__ tb2,
    const float* __restrict__ tw3, const float* __restrict__ tb3,
    float* __restrict__ tower) {
  __shared__ unsigned short X[16 * XKP];        // x tile, bf16
  __shared__ unsigned short T1s[8 * 16 * T1N];  // per-wave t1 staging, bf16
  __shared__ float          T2s[8 * 16 * 16];   // per-wave t2 staging, fp32
  const int b0  = blockIdx.x * 16;
  const int tid = threadIdx.x;
  for (int idx = tid; idx < 16 * XKP; idx += 256)
    X[idx] = xbuf[(size_t)b0 * XKP + idx];
  __syncthreads();

  const int wave = tid >> 5, lane = tid & 31;
  const int hi = lane >> 4, ln = lane & 15;

  for (int ci = 0; ci < 2; ++ci) {
    const int c = wave + ci * 8;
    // ---- layer 1: [16 x 96] x [96 x 32]
    v8f a1[2];
#pragma unroll
    for (int t = 0; t < 2; ++t) {
      float bias = tb1[c * T1N + t * 16 + ln];
#pragma unroll
      for (int r = 0; r < 8; ++r) a1[t][r] = bias;
    }
    for (int kc = 0; kc < XKP; kc += 32) {
      v16bf af = frag_ld(X, XKP, ln, kc, lane);
#pragma unroll
      for (int t = 0; t < 2; ++t) {
        v16bf bf = frag_ld(T1p, XKP, c * T1N + t * 16 + ln, kc, lane);
        a1[t] = WMMA_BF16(af, bf, a1[t]);
      }
    }
#pragma unroll
    for (int t = 0; t < 2; ++t)
#pragma unroll
      for (int r = 0; r < 8; ++r) {
        float v = a1[t][r];
        v = v > 0.f ? v : 0.f;
        T1s[(wave * 16 + r + 8 * hi) * T1N + t * 16 + ln] = f2bf(v);
      }
    // ---- layer 2: [16 x 32] x [32 x 16(pad of 10)], single WMMA
    v8f a2;
    {
      float bias = (ln < T2N) ? tb2[c * T2N + ln] : 0.f;
#pragma unroll
      for (int r = 0; r < 8; ++r) a2[r] = bias;
    }
    {
      v16bf af = frag_ld(T1s, T1N, wave * 16 + ln, 0, lane);
      v16bf bf = frag_ld(T2p, T1N, c * T2NP + ln,  0, lane);
      a2 = WMMA_BF16(af, bf, a2);
    }
#pragma unroll
    for (int r = 0; r < 8; ++r) {
      float v = a2[r];
      T2s[(wave * 16 + r + 8 * hi) * 16 + ln] = v > 0.f ? v : 0.f;
    }
    // ---- layer 3 + sigmoid (per-row dot over 10), lanes 0..15 only
    if (lane < 16) {
      float s = tb3[c];
#pragma unroll
      for (int o = 0; o < T2N; ++o)
        s += T2s[(wave * 16 + lane) * 16 + o] * tw3[c * T2N + o];
      tower[(size_t)(b0 + lane) * NC + c] = 1.f / (1.f + __expf(-s));
    }
  }
}

// ---------------------------------------------------------------------------
// Kernel 3: Student-t soft assignment (alpha=1 -> q = 1/(1+d2)), normalize,
// final weighted sum. One thread per batch row.
__global__ __launch_bounds__(256) void bk_final(
    const float* __restrict__ urep, const float* __restrict__ centers,
    const float* __restrict__ tower, float* __restrict__ q_out,
    float* __restrict__ s_out) {
  __shared__ float C[NC * DREP];
  const int tid = threadIdx.x;
  for (int i = tid; i < NC * DREP; i += 256) C[i] = centers[i];
  __syncthreads();
  const int b = blockIdx.x * 256 + tid;
  const float* ur = urep + (size_t)b * DREP;
  float u[DREP];
#pragma unroll
  for (int i = 0; i < DREP; i += 4) {
    float4 t = *(const float4*)(ur + i);
    u[i] = t.x; u[i + 1] = t.y; u[i + 2] = t.z; u[i + 3] = t.w;
  }
  float q[NC], qs = 0.f;
#pragma unroll
  for (int c = 0; c < NC; ++c) {
    float d2 = 0.f;
#pragma unroll
    for (int i = 0; i < DREP; ++i) {
      float d = u[i] - C[c * DREP + i];
      d2 += d * d;
    }
    q[c] = 1.f / (1.f + d2);
    qs += q[c];
  }
  const float inv = 1.f / qs;
  float s = 0.f;
#pragma unroll
  for (int c = 0; c < NC; ++c) {
    float qq = q[c] * inv;
    q_out[(size_t)b * NC + c] = qq;
    s += tower[(size_t)b * NC + c] * qq;
  }
  s_out[b] = s;
}

// ---------------------------------------------------------------------------
extern "C" void kernel_launch(void* const* d_in, const int* in_sizes, int n_in,
                              void* d_out, int out_size, void* d_ws, size_t ws_size,
                              hipStream_t stream) {
  const int*   u_feat     = (const int*)  d_in[0];
  const int*   i_feat     = (const int*)  d_in[1];
  const float* user_table = (const float*)d_in[2];
  const float* w_rep1     = (const float*)d_in[3];
  const float* b_rep1     = (const float*)d_in[4];
  const float* w_rep2     = (const float*)d_in[5];
  const float* b_rep2     = (const float*)d_in[6];
  const float* item_table = (const float*)d_in[7];
  const float* tw1        = (const float*)d_in[8];
  const float* tb1        = (const float*)d_in[9];
  const float* tw2        = (const float*)d_in[10];
  const float* tb2        = (const float*)d_in[11];
  const float* tw3        = (const float*)d_in[12];
  const float* tb3        = (const float*)d_in[13];
  const float* centers    = (const float*)d_in[14];

  const int B = in_sizes[1];   // 65536

  // workspace layout (bytes)
  char* ws = (char*)d_ws;
  unsigned short* W1p  = (unsigned short*)(ws);                       // 256*224*2
  unsigned short* W2p  = (unsigned short*)(ws + 114688);              // 64*256*2
  unsigned short* T1p  = (unsigned short*)(ws + 147456);              // 16*32*96*2
  unsigned short* T2p  = (unsigned short*)(ws + 245760);              // 16*16*32*2
  unsigned short* xbuf = (unsigned short*)(ws + 262144);              // B*96*2
  float*          urep = (float*)        (ws + 262144 + (size_t)B * XKP * 2);
  float*          twr  = (float*)        (ws + 262144 + (size_t)B * XKP * 2
                                              + (size_t)B * DREP * 4);

  float* q_out = (float*)d_out;
  float* s_out = q_out + (size_t)B * NC;

  const int prepElems = H1 * K1P + DREP * H1 + NC * T1N * XKP + NC * T2NP * T1N;
  bk_prep<<<(prepElems + 255) / 256, 256, 0, stream>>>(
      w_rep1, w_rep2, tw1, tw2, W1p, W2p, T1p, T2p);

  bk_user<<<B / 32, 256, 0, stream>>>(
      u_feat, i_feat, user_table, b_rep1, b_rep2, item_table,
      W1p, W2p, xbuf, urep);

  bk_tower<<<B / 16, 256, 0, stream>>>(
      xbuf, T1p, T2p, tb1, tb2, tw3, tb3, twr);

  bk_final<<<B / 256, 256, 0, stream>>>(urep, centers, twr, q_out, s_out);
}
